// CrossAttentionModule_33054068310284
// MI455X (gfx1250) — matrix operand order repaired and checked by reading, not compiled, not run on hardware
//
#include <hip/hip_runtime.h>

// ---------------------------------------------------------------------------
// CDNA5 (gfx1250) cross-attention, bf16 WMMA, single-pass no-max softmax.
//   Q  = img  @ W_Q^T           [8192,1024]  bf16 (ws)
//   K  = text @ W_K^T           [8192,1024]  bf16 (ws)
//   Vt = W_V  @ text^T          [1024,8192]  bf16 (ws)  (transposed V)
//   out[m,:] = (sum_j exp(q_m.k_j/32) * V[j,:]) / (sum_j exp(q_m.k_j/32))
// ---------------------------------------------------------------------------

typedef __bf16 bf16;
typedef __attribute__((ext_vector_type(16))) __bf16 v16bf;
typedef __attribute__((ext_vector_type(8)))  __bf16 v8bf;
typedef __attribute__((ext_vector_type(8)))  float  v8f;
typedef __attribute__((ext_vector_type(4)))  float  v4f;
typedef __attribute__((vector_size(4 * sizeof(int)))) int v4i;
typedef __attribute__((address_space(3))) v4i lds_v4i;

#define N_IMG 8192
#define N_TXT 8192
#define DIM   1024

static __device__ __forceinline__ v8f wmma_bf16(v16bf a, v16bf b, v8f c) {
  // D = A(16x32) * B(32x16) + C, f32 accumulate
  return __builtin_amdgcn_wmma_f32_16x16x32_bf16(false, a, false, b, (short)0, c,
                                                 false, false);
}

static __device__ __forceinline__ v16bf pack16(v8bf lo, v8bf hi) {
  return __builtin_shufflevector(lo, hi, 0, 1, 2, 3, 4, 5, 6, 7,
                                 8, 9, 10, 11, 12, 13, 14, 15);
}

// A fragment (16x32, MxK) from row-major bf16, leading dim ld.
// lane L holds row (L&15); K chunks base=(L>>4)*8: [base..base+7],[base+16..base+23]
static __device__ __forceinline__ v16bf frag_a_bf16(const bf16* M, int ld,
                                                    int row, int k0, int lane) {
  const bf16* p = M + (size_t)(row + (lane & 15)) * ld + k0 + ((lane >> 4) << 3);
  return pack16(*(const v8bf*)p, *(const v8bf*)(p + 16));
}

// B fragment (32x16, KxN) for C = A * W^T : B[k][n] = W[n][k].
// lane L holds column n=(L&15) (= W row n); K chunks base=(L>>4)*8: [base..base+7],[base+16..base+23]
static __device__ __forceinline__ v16bf frag_b_bf16(const bf16* M, int ld,
                                                    int row, int k0, int lane) {
  const bf16* p = M + (size_t)(row + (lane & 15)) * ld + k0 + ((lane >> 4) << 3);
  return pack16(*(const v8bf*)p, *(const v8bf*)(p + 16));
}

// Same fragments but gathered from row-major f32 with on-the-fly bf16 convert.
static __device__ __forceinline__ v16bf frag_a_f32(const float* M, int ld,
                                                   int row, int k0, int lane) {
  const float* p = M + (size_t)(row + (lane & 15)) * ld + k0 + ((lane >> 4) << 3);
  v4f x0 = *(const v4f*)p, x1 = *(const v4f*)(p + 4);
  v4f x2 = *(const v4f*)(p + 16), x3 = *(const v4f*)(p + 20);
  v16bf a;
#pragma unroll
  for (int i = 0; i < 4; ++i) {
    a[i] = (bf16)x0[i]; a[4 + i] = (bf16)x1[i];
    a[8 + i] = (bf16)x2[i]; a[12 + i] = (bf16)x3[i];
  }
  return a;
}

static __device__ __forceinline__ v16bf frag_b_f32(const float* M, int ld,
                                                   int row, int k0, int lane) {
  const float* p = M + (size_t)(row + (lane & 15)) * ld + k0 + ((lane >> 4) << 3);
  v4f x0 = *(const v4f*)p, x1 = *(const v4f*)(p + 4);
  v4f x2 = *(const v4f*)(p + 16), x3 = *(const v4f*)(p + 20);
  v16bf b;
#pragma unroll
  for (int i = 0; i < 4; ++i) {
    b[i] = (bf16)x0[i]; b[4 + i] = (bf16)x1[i];
    b[8 + i] = (bf16)x2[i]; b[12 + i] = (bf16)x3[i];
  }
  return b;
}

// ---------------------------------------------------------------------------
// Projection GEMM: C[M,N] (bf16, row-major) = A[M,K] @ B[N,K]^T  (f32 inputs)
// 256 threads = 8 waves; 128x128 tile; wave w -> rows (w&3)*32, cols (w>>2)*64.
// ---------------------------------------------------------------------------
__global__ __launch_bounds__(256) void proj_gemm_bf16(
    const float* __restrict__ A, const float* __restrict__ B,
    bf16* __restrict__ C, int M, int N, int K) {
  const int lane = threadIdx.x & 31;
  const int w = threadIdx.x >> 5;
  const int mB = blockIdx.y * 128 + (w & 3) * 32;
  const int nB = blockIdx.x * 128 + (w >> 2) * 64;

  v8f acc[2][4] = {};
  for (int k0 = 0; k0 < K; k0 += 32) {
    v16bf a0 = frag_a_f32(A, K, mB, k0, lane);
    v16bf a1 = frag_a_f32(A, K, mB + 16, k0, lane);
#pragma unroll
    for (int nt = 0; nt < 4; ++nt) {
      v16bf b = frag_b_f32(B, K, nB + nt * 16, k0, lane);
      acc[0][nt] = wmma_bf16(a0, b, acc[0][nt]);
      acc[1][nt] = wmma_bf16(a1, b, acc[1][nt]);
    }
  }
#pragma unroll
  for (int mt = 0; mt < 2; ++mt)
#pragma unroll
    for (int nt = 0; nt < 4; ++nt)
#pragma unroll
      for (int r = 0; r < 8; ++r) {
        int row = mB + mt * 16 + r + ((lane >> 4) << 3);
        int col = nB + nt * 16 + (lane & 15);
        C[(size_t)row * N + col] = (bf16)acc[mt][nt][r];
      }
}

// ---------------------------------------------------------------------------
// Fused attention: 256 blocks x 256 threads (8 waves). M-block = 32 rows.
// LDS: Qs[32][1032] bf16 (padded), Ps[32][72] bf16, ls[32] f32.
// Wave w: computes S tile (mi=w>>2, ni=w&3) then owns dout slice [w*128,w*128+128).
// Q block staged global->LDS with CDNA5 async-to-LDS when available (ASYNCcnt).
// ---------------------------------------------------------------------------
#define LDQ 1032
#define LDP 72
#define ATTN_LDS_BYTES (32 * LDQ * 2 + 32 * LDP * 2 + 32 * 4)

#if __has_builtin(__builtin_amdgcn_global_load_async_to_lds_b128)
#define HAVE_ASYNC_LDS 1
#else
#define HAVE_ASYNC_LDS 0
#endif

static __device__ __forceinline__ void wait_asynccnt0() {
#if __has_builtin(__builtin_amdgcn_s_wait_asynccnt)
  __builtin_amdgcn_s_wait_asynccnt(0);
#else
  asm volatile("s_wait_asynccnt 0" ::: "memory");
#endif
}

__global__ __launch_bounds__(256) void attn_fused(
    const bf16* __restrict__ Qb, const bf16* __restrict__ Kb,
    const bf16* __restrict__ Vt, float* __restrict__ Out) {
  extern __shared__ __align__(16) char smem[];
  bf16* Qs = (bf16*)smem;                            // 32 x 1032
  bf16* Ps = (bf16*)(smem + 32 * LDQ * 2);           // 32 x 72
  float* ls = (float*)(smem + 32 * LDQ * 2 + 32 * LDP * 2);  // 32

  const int tid = threadIdx.x;
  const int lane = tid & 31;
  const int w = tid >> 5;
  const int m0 = blockIdx.x * 32;

  // Stage this block's Q rows (bf16) into padded LDS, zero the denominators.
#if HAVE_ASYNC_LDS
  for (int c = tid; c < 32 * 128; c += 256) {   // 8-element (16B) chunks
    int row = c >> 7, kc = (c & 127) << 3;
    __builtin_amdgcn_global_load_async_to_lds_b128(
        (v4i*)(Qb + (size_t)(m0 + row) * DIM + kc),
        (lds_v4i*)(Qs + row * LDQ + kc),
        0, 0);
  }
  if (tid < 32) ls[tid] = 0.0f;
  wait_asynccnt0();
#else
  for (int c = tid; c < 32 * 128; c += 256) {   // 8-element chunks
    int row = c >> 7, kc = (c & 127) << 3;
    *(v8bf*)(Qs + row * LDQ + kc) =
        *(const v8bf*)(Qb + (size_t)(m0 + row) * DIM + kc);
  }
  if (tid < 32) ls[tid] = 0.0f;
#endif
  __syncthreads();

  const int mi = w >> 2;        // 0..1 : 16-row slice of S
  const int ni = w & 3;         // 0..3 : 16-col slice of S
  const int doutB = w * 128;    // wave-private 128-wide output slice
  const float scale = 0.03125f; // 1/sqrt(1024)

  v8f acc[2][8] = {};           // 32 x 128 f32 output accumulator per wave

  for (int t = 0; t < N_TXT; t += 64) {
    // ---- S tile: 16x16 scores over K=1024 (32 WMMAs) ----
    v8f s = {};
    for (int k0 = 0; k0 < DIM; k0 += 32) {
      v16bf aq = frag_a_bf16(Qs, LDQ, mi * 16, k0, lane);          // LDS
      v16bf bk = frag_b_bf16(Kb, DIM, t + ni * 16, k0, lane);      // global
      s = wmma_bf16(aq, bk, s);
    }
    if (t + 64 < N_TXT) {  // hint next K tile toward the caches
      __builtin_prefetch(Kb + (size_t)(t + 64 + ni * 16 + (lane & 15)) * DIM +
                             ((lane >> 4) << 4), 0, 0);
    }

    // ---- exp, per-row denominators (16-lane shuffles + ds_add), P -> LDS ----
#pragma unroll
    for (int r = 0; r < 8; ++r) {
      float p = __expf(s[r] * scale);
      float rs = p;
      rs += __shfl_xor(rs, 1, 16);
      rs += __shfl_xor(rs, 2, 16);
      rs += __shfl_xor(rs, 4, 16);
      rs += __shfl_xor(rs, 8, 16);
      int mrow = mi * 16 + r + ((lane >> 4) << 3);
      if ((lane & 15) == 0) atomicAdd(&ls[mrow], rs);
      Ps[mrow * LDP + ni * 16 + (lane & 15)] = (bf16)p;
    }
    __syncthreads();

    // ---- O += P(32x64) @ Vt-slice(64 x 128): 32 WMMAs ----
#pragma unroll
    for (int nt = 0; nt < 8; ++nt) {
      v16bf b0 = frag_b_bf16(Vt, N_TXT, doutB + nt * 16, t, lane);
      v16bf b1 = frag_b_bf16(Vt, N_TXT, doutB + nt * 16, t + 32, lane);
#pragma unroll
      for (int mt = 0; mt < 2; ++mt) {
        v16bf a0 = frag_a_bf16(Ps, LDP, mt * 16, 0, lane);
        v16bf a1 = frag_a_bf16(Ps, LDP, mt * 16, 32, lane);
        acc[mt][nt] = wmma_bf16(a0, b0, acc[mt][nt]);
        acc[mt][nt] = wmma_bf16(a1, b1, acc[mt][nt]);
      }
    }
    __syncthreads();  // before next iteration overwrites Ps
  }

  // ---- finalize: out = O / l ----
#pragma unroll
  for (int mt = 0; mt < 2; ++mt)
#pragma unroll
    for (int r = 0; r < 8; ++r) {
      int mrow = mt * 16 + r + ((lane >> 4) << 3);
      float inv = 1.0f / ls[mrow];
#pragma unroll
      for (int nt = 0; nt < 8; ++nt) {
        Out[(size_t)(m0 + mrow) * DIM + doutB + nt * 16 + (lane & 15)] =
            acc[mt][nt][r] * inv;
      }
    }
}

// ---------------------------------------------------------------------------
extern "C" void kernel_launch(void* const* d_in, const int* in_sizes, int n_in,
                              void* d_out, int out_size, void* d_ws,
                              size_t ws_size, hipStream_t stream) {
  const float* img  = (const float*)d_in[0];   // [8192,1024]
  const float* text = (const float*)d_in[1];   // [8192,1024]
  const float* W_Q  = (const float*)d_in[2];   // [1024,1024]
  const float* W_K  = (const float*)d_in[3];   // [1024,1024]
  const float* W_V  = (const float*)d_in[4];   // [1024,1024]
  float* out = (float*)d_out;                  // [8192,1024]

  bf16* Qb = (bf16*)d_ws;                      // 16 MB
  bf16* Kb = Qb + (size_t)N_IMG * DIM;         // 16 MB
  bf16* Vt = Kb + (size_t)N_TXT * DIM;         // 16 MB  (transposed V)

  // Q = img @ W_Q^T, K = text @ W_K^T  (row-major bf16)
  proj_gemm_bf16<<<dim3(DIM / 128, N_IMG / 128), 256, 0, stream>>>(
      img, W_Q, Qb, N_IMG, DIM, DIM);
  proj_gemm_bf16<<<dim3(DIM / 128, N_TXT / 128), 256, 0, stream>>>(
      text, W_K, Kb, N_TXT, DIM, DIM);
  // Vt = W_V @ text^T  (same NT-GEMM, operands swapped -> transposed output)
  proj_gemm_bf16<<<dim3(N_TXT / 128, DIM / 128), 256, 0, stream>>>(
      W_V, text, Vt, DIM, N_TXT, DIM);

  attn_fused<<<N_IMG / 32, 256, ATTN_LDS_BYTES, stream>>>(Qb, Kb, Vt, out);
}
